// RSAGE_GAT_22333829939349
// MI455X (gfx1250) — matrix operand order repaired and checked by reading, not compiled, not run on hardware
//
#include <hip/hip_runtime.h>
#include <hip/hip_bf16.h>
#include <cstddef>

// ---------------------------------------------------------------------------
// Types for CDNA5 WMMA
// ---------------------------------------------------------------------------
typedef __attribute__((ext_vector_type(16))) __bf16 v16bf;
typedef __attribute__((ext_vector_type(8)))  float  v8f;

#define DHID 128
#define NHEAD 4
#define DPH 32

// ---------------------------------------------------------------------------
// Utility kernels
// ---------------------------------------------------------------------------
__global__ void fill_f32(float* __restrict__ p, float v, size_t n) {
    size_t i = (size_t)blockIdx.x * blockDim.x + threadIdx.x;
    if (i < n) p[i] = v;
}

// ---------------------------------------------------------------------------
// WMMA fragment helpers (CDNA5 16x16x32 bf16 layouts, wave32)
//   A 16x32: lane = row (lane&15); e<8 -> K = kb+e, e>=8 -> K = kb+16+(e-8),
//            kb = (lane>=16) ? 8 : 0
//   B 32x16: lane = column (lane&15); same K mapping as A
//   C 16x16: col = lane&15, row = v + ((lane>=16) ? 8 : 0)
// ---------------------------------------------------------------------------
__device__ __forceinline__ v16bf load_a_frag(const float* __restrict__ Abase,
                                             int kc, float scale) {
    v16bf a;
#pragma unroll
    for (int e = 0; e < 8; ++e) {
        a[e]     = (__bf16)(Abase[kc + e] * scale);
        a[e + 8] = (__bf16)(Abase[kc + 16 + e] * scale);
    }
    return a;
}

template <int WP>
__device__ __forceinline__ v16bf load_b_frag(const __bf16* __restrict__ lds,
                                             int kc, int lane, int t) {
    const __bf16* Bcol = lds + (size_t)(kc + ((lane >= 16) ? 8 : 0)) * WP + t * 16 + (lane & 15);
    v16bf b;
#pragma unroll
    for (int e = 0; e < 8; ++e) {
        b[e]     = Bcol[(size_t)e * WP];
        b[e + 8] = Bcol[(size_t)(16 + e) * WP];
    }
    return b;
}

// ---------------------------------------------------------------------------
// Fused SAGE GEMM:
//   acc[m, :] += hin[m, :] @ Wself  +  (hn[m, :] / max(deg[m],1)) @ Wneigh + b
// One block = 128 rows, 8 waves; both 128x128 weight matrices staged in LDS
// as bf16; two K-phases accumulate into the same WMMA accumulators; single
// read-modify-write epilogue on acc.
// ---------------------------------------------------------------------------
__global__ void __launch_bounds__(256)
sage_gemm_fused(const float* __restrict__ hin, const float* __restrict__ hn,
                const float* __restrict__ deg,
                const float* __restrict__ Wself, const float* __restrict__ Wneigh,
                const float* __restrict__ bias, float* __restrict__ acc, int nrows) {
    constexpr int WP = DHID + 8;                 // padded LDS row (bank-conflict dodge)
    __shared__ __bf16 ldsWs[DHID * WP];
    __shared__ __bf16 ldsWn[DHID * WP];

    const int tid = threadIdx.x;
    for (int idx = tid; idx < DHID * DHID; idx += 256) {
        int k = idx >> 7;
        int n = idx & 127;
        ldsWs[k * WP + n] = (__bf16)Wself[idx];
        ldsWn[k * WP + n] = (__bf16)Wneigh[idx];
    }
    __syncthreads();

    const int wave = tid >> 5;
    const int lane = tid & 31;
    const int m0   = blockIdx.x * 128 + wave * 16;
    if (m0 >= nrows) return;                     // wave-uniform: EXEC stays all-1s

    v8f c[8];
#pragma unroll
    for (int t = 0; t < 8; ++t) {
        v8f z = {0.f, 0.f, 0.f, 0.f, 0.f, 0.f, 0.f, 0.f};
        c[t] = z;
    }

    int arow = m0 + (lane & 15);
    if (arow >= nrows) arow = nrows - 1;
    const int kb = (lane >= 16) ? 8 : 0;

    // ---- Phase 1: self term ----
    {
        const float* Abase = hin + (size_t)arow * DHID + kb;
#pragma unroll
        for (int kc = 0; kc < DHID; kc += 32) {
            v16bf a = load_a_frag(Abase, kc, 1.0f);
#pragma unroll
            for (int t = 0; t < 8; ++t) {
                v16bf b = load_b_frag<WP>(ldsWs, kc, lane, t);
                c[t] = __builtin_amdgcn_wmma_f32_16x16x32_bf16(
                    false, a, false, b, (short)0, c[t], false, false);
            }
        }
    }
    // ---- Phase 2: neighbor term, degree normalization folded into A load ----
    {
        const float scale = 1.0f / fmaxf(deg[arow], 1.0f);
        const float* Abase = hn + (size_t)arow * DHID + kb;
#pragma unroll
        for (int kc = 0; kc < DHID; kc += 32) {
            v16bf a = load_a_frag(Abase, kc, scale);
#pragma unroll
            for (int t = 0; t < 8; ++t) {
                v16bf b = load_b_frag<WP>(ldsWn, kc, lane, t);
                c[t] = __builtin_amdgcn_wmma_f32_16x16x32_bf16(
                    false, a, false, b, (short)0, c[t], false, false);
            }
        }
    }

    // ---- Epilogue: single RMW on acc, bias folded in ----
    const int colbase = lane & 15;
    const int mrow0   = m0 + ((lane >= 16) ? 8 : 0);
#pragma unroll
    for (int t = 0; t < 8; ++t) {
        int   col = t * 16 + colbase;
        float bv  = bias[col];
#pragma unroll
        for (int v = 0; v < 8; ++v) {
            int m = mrow0 + v;
            if (m < nrows) {
                size_t o = (size_t)m * DHID + col;
                acc[o] += c[t][v] + bv;
            }
        }
    }
}

// ---------------------------------------------------------------------------
// Generic WMMA GEMM:  C[nrows x OUTC] (+)= A[nrows x 128] @ W[128 x OUTC] (+b)
// ---------------------------------------------------------------------------
template <int NTILES>
__global__ void __launch_bounds__(256)
gemm_wmma_bf16(const float* __restrict__ A, const float* __restrict__ W,
               const float* __restrict__ bias, float* __restrict__ C,
               int nrows, int accumulate) {
    constexpr int OUTC = NTILES * 16;
    constexpr int WP   = OUTC + 8;
    __shared__ __bf16 ldsW[128 * WP];

    const int tid = threadIdx.x;
    for (int idx = tid; idx < 128 * OUTC; idx += 256) {
        int k = idx / OUTC;
        int n = idx - k * OUTC;
        ldsW[k * WP + n] = (__bf16)W[idx];
    }
    __syncthreads();

    const int wave = tid >> 5;
    const int lane = tid & 31;
    const int m0   = blockIdx.x * 128 + wave * 16;
    if (m0 >= nrows) return;

    v8f c[NTILES];
#pragma unroll
    for (int t = 0; t < NTILES; ++t) {
        v8f z = {0.f, 0.f, 0.f, 0.f, 0.f, 0.f, 0.f, 0.f};
        c[t] = z;
    }

    int arow = m0 + (lane & 15);
    if (arow >= nrows) arow = nrows - 1;
    const float* Abase = A + (size_t)arow * 128 + ((lane >= 16) ? 8 : 0);

#pragma unroll
    for (int kc = 0; kc < 128; kc += 32) {
        v16bf a = load_a_frag(Abase, kc, 1.0f);
#pragma unroll
        for (int t = 0; t < NTILES; ++t) {
            v16bf b = load_b_frag<WP>(ldsW, kc, lane, t);
            c[t] = __builtin_amdgcn_wmma_f32_16x16x32_bf16(
                false, a, false, b, (short)0, c[t], false, false);
        }
    }

    const int colbase = lane & 15;
    const int mrow0   = m0 + ((lane >= 16) ? 8 : 0);
#pragma unroll
    for (int t = 0; t < NTILES; ++t) {
        int   col = t * 16 + colbase;
        float bv  = bias ? bias[col] : 0.f;
#pragma unroll
        for (int v = 0; v < 8; ++v) {
            int m = mrow0 + v;
            if (m < nrows) {
                size_t o   = (size_t)m * OUTC + col;
                float  val = c[t][v] + bv;
                C[o] = accumulate ? (C[o] + val) : val;
            }
        }
    }
}

// ---------------------------------------------------------------------------
// SAGE edge aggregation: one wave per edge; lane carries 4 of 128 floats.
// hn[dst] += h[src]; deg[dst] += 1
// ---------------------------------------------------------------------------
__global__ void sage_agg(const float* __restrict__ h, const int* __restrict__ src,
                         const int* __restrict__ dst, float* __restrict__ hn,
                         float* __restrict__ deg, int nedges) {
    int wid  = (int)(((size_t)blockIdx.x * blockDim.x + threadIdx.x) >> 5);
    int lane = threadIdx.x & 31;
    if (wid >= nedges) return;
    int s = src[wid], d = dst[wid];
    const float4 v = ((const float4*)(h + (size_t)s * DHID))[lane];
    float* out = hn + (size_t)d * DHID + lane * 4;
    atomicAdd(out + 0, v.x);
    atomicAdd(out + 1, v.y);
    atomicAdd(out + 2, v.z);
    atomicAdd(out + 3, v.w);
    if (lane == 0) atomicAdd(deg + d, 1.0f);
}

// h = leaky_relu(acc / 3, 0.01)
__global__ void mean_lrelu(const float* __restrict__ acc, float* __restrict__ h,
                           size_t total) {
    size_t i = (size_t)blockIdx.x * blockDim.x + threadIdx.x;
    if (i >= total) return;
    float v = acc[i] * (1.0f / 3.0f);
    h[i] = (v > 0.f) ? v : 0.01f * v;
}

// ---------------------------------------------------------------------------
// GAT kernels
// ---------------------------------------------------------------------------
__global__ void gat_el_er(const float* __restrict__ f, const float* __restrict__ al,
                          const float* __restrict__ ar, float* __restrict__ el,
                          float* __restrict__ er, int n) {
    int idx = blockIdx.x * blockDim.x + threadIdx.x;   // node*H + head
    if (idx >= n * NHEAD) return;
    int node = idx >> 2, hh = idx & 3;
    const float* fp  = f + (size_t)node * DHID + hh * DPH;
    const float* alp = al + hh * DPH;
    const float* arp = ar + hh * DPH;
    float sl = 0.f, sr = 0.f;
#pragma unroll 8
    for (int d = 0; d < DPH; ++d) {
        sl += fp[d] * alp[d];
        sr += fp[d] * arp[d];
    }
    el[idx] = sl;
    er[idx] = sr;
}

__device__ __forceinline__ void atomicMaxF(float* addr, float val) {
    unsigned int* ai = (unsigned int*)addr;
    unsigned int old = *ai;
    while (__uint_as_float(old) < val) {
        unsigned int assumed = old;
        old = atomicCAS(ai, assumed, __float_as_uint(val));
        if (old == assumed) break;
    }
}

__global__ void gat_edge_max(const float* __restrict__ el, const float* __restrict__ er,
                             const int* __restrict__ src, const int* __restrict__ dst,
                             float* __restrict__ m, int nedges) {
    int idx = (int)((size_t)blockIdx.x * blockDim.x + threadIdx.x);
    if (idx >= nedges * NHEAD) return;
    int e = idx >> 2, hh = idx & 3;
    int s = src[e], d = dst[e];
    float x = el[s * NHEAD + hh] + er[d * NHEAD + hh];
    x = (x > 0.f) ? x : 0.2f * x;
    atomicMaxF(m + d * NHEAD + hh, x);
}

__global__ void gat_edge_sum(const float* __restrict__ el, const float* __restrict__ er,
                             const int* __restrict__ src, const int* __restrict__ dst,
                             const float* __restrict__ m, float* __restrict__ ssum,
                             int nedges) {
    int idx = (int)((size_t)blockIdx.x * blockDim.x + threadIdx.x);
    if (idx >= nedges * NHEAD) return;
    int e = idx >> 2, hh = idx & 3;
    int s = src[e], d = dst[e];
    float x = el[s * NHEAD + hh] + er[d * NHEAD + hh];
    x = (x > 0.f) ? x : 0.2f * x;
    float ex = __expf(x - m[d * NHEAD + hh]);
    atomicAdd(ssum + d * NHEAD + hh, ex);
}

// one wave per edge: lane covers dims [lane*4, lane*4+4) -> head = lane>>3
__global__ void gat_scatter(const float* __restrict__ f, const float* __restrict__ el,
                            const float* __restrict__ er, const int* __restrict__ src,
                            const int* __restrict__ dst, const float* __restrict__ m,
                            const float* __restrict__ ssum, float* __restrict__ gacc,
                            int nedges) {
    int wid  = (int)(((size_t)blockIdx.x * blockDim.x + threadIdx.x) >> 5);
    int lane = threadIdx.x & 31;
    if (wid >= nedges) return;
    int s = src[wid], d = dst[wid];
    int hh = lane >> 3;
    float x = el[s * NHEAD + hh] + er[d * NHEAD + hh];
    x = (x > 0.f) ? x : 0.2f * x;
    float ex    = __expf(x - m[d * NHEAD + hh]);
    float alpha = ex / fmaxf(ssum[d * NHEAD + hh], 1e-9f);
    const float4 fv = ((const float4*)(f + (size_t)s * DHID))[lane];
    float* out = gacc + (size_t)d * DHID + lane * 4;
    atomicAdd(out + 0, alpha * fv.x);
    atomicAdd(out + 1, alpha * fv.y);
    atomicAdd(out + 2, alpha * fv.z);
    atomicAdd(out + 3, alpha * fv.w);
}

// h = gacc/3 + mean_r gat_b[r]
__global__ void gat_finalize(const float* __restrict__ gacc, const float* __restrict__ gb,
                             float* __restrict__ h, size_t total) {
    size_t i = (size_t)blockIdx.x * blockDim.x + threadIdx.x;
    if (i >= total) return;
    int c = (int)(i & 127);
    float bm = (gb[c] + gb[DHID + c] + gb[2 * DHID + c]) * (1.0f / 3.0f);
    h[i] = gacc[i] * (1.0f / 3.0f) + bm;
}

// ---------------------------------------------------------------------------
// Host orchestration
// ---------------------------------------------------------------------------
static inline int cdiv(long long a, long long b) { return (int)((a + b - 1) / b); }

extern "C" void kernel_launch(void* const* d_in, const int* in_sizes, int n_in,
                              void* d_out, int out_size, void* d_ws, size_t ws_size,
                              hipStream_t stream) {
    const float* feat         = (const float*)d_in[0];
    const int*   src          = (const int*)d_in[1];
    const int*   dst          = (const int*)d_in[2];
    const float* sage0_Wself  = (const float*)d_in[3];
    const float* sage0_Wneigh = (const float*)d_in[4];
    const float* sage0_b      = (const float*)d_in[5];
    const float* sage_Wself   = (const float*)d_in[6];
    const float* sage_Wneigh  = (const float*)d_in[7];
    const float* sage_b       = (const float*)d_in[8];
    const float* gat_W        = (const float*)d_in[9];
    const float* gat_al       = (const float*)d_in[10];
    const float* gat_ar       = (const float*)d_in[11];
    const float* gat_b        = (const float*)d_in[12];
    const float* lin_W        = (const float*)d_in[13];
    const float* lin_b        = (const float*)d_in[14];
    float* out = (float*)d_out;

    const int N = in_sizes[0] / DHID;   // 100000
    const int R = 3;
    const int E = in_sizes[1] / R;      // 800000
    const size_t NF = (size_t)N * DHID;

    float* ws   = (float*)d_ws;
    float* acc  = ws;                 // N*128
    float* hn   = ws + NF;            // N*128  (also GAT f buffer)
    float* hcur = ws + 2 * NF;        // N*128
    float* deg  = ws + 3 * NF;        // N
    float* el   = deg + N;            // N*4
    float* er   = el + (size_t)N * 4; // N*4
    float* mx   = er + (size_t)N * 4; // N*4
    float* ssum = mx + (size_t)N * 4; // N*4

    const int TB = 256;
    const int gElem  = cdiv(NF, TB);
    const int gN     = cdiv(N, TB);
    const int gN4    = cdiv((long long)N * 4, TB);
    const int gEdgeW = cdiv((long long)E * 32, TB);
    const int gEdge4 = cdiv((long long)E * 4, TB);
    const int gGemm  = cdiv(N, 128);

    const size_t WSZ = (size_t)DHID * DHID;

    // ---------------- SAGE layers 0..3 ----------------
    const float* hin = feat;
    for (int layer = 0; layer < 4; ++layer) {
        fill_f32<<<gElem, TB, 0, stream>>>(acc, 0.f, NF);
        for (int r = 0; r < R; ++r) {
            const float *Ws, *Wn, *bb;
            if (layer == 0) {
                Ws = sage0_Wself  + (size_t)r * WSZ;
                Wn = sage0_Wneigh + (size_t)r * WSZ;
                bb = sage0_b      + (size_t)r * DHID;
            } else {
                size_t o = ((size_t)(layer - 1) * R + r);
                Ws = sage_Wself  + o * WSZ;
                Wn = sage_Wneigh + o * WSZ;
                bb = sage_b      + o * DHID;
            }
            const int* sr = src + (size_t)r * E;
            const int* dr = dst + (size_t)r * E;
            fill_f32<<<gElem, TB, 0, stream>>>(hn, 0.f, NF);
            fill_f32<<<gN, TB, 0, stream>>>(deg, 0.f, (size_t)N);
            sage_agg<<<gEdgeW, TB, 0, stream>>>(hin, sr, dr, hn, deg, E);
            // acc += hin@Wself + (hn/deg)@Wneigh + b   (one fused WMMA kernel)
            sage_gemm_fused<<<gGemm, TB, 0, stream>>>(hin, hn, deg, Ws, Wn, bb, acc, N);
        }
        mean_lrelu<<<gElem, TB, 0, stream>>>(acc, hcur, NF);
        hin = hcur;
    }

    // ---------------- GAT layer ----------------
    fill_f32<<<gElem, TB, 0, stream>>>(acc, 0.f, NF);   // gacc
    for (int r = 0; r < R; ++r) {
        const int* sr = src + (size_t)r * E;
        const int* dr = dst + (size_t)r * E;
        gemm_wmma_bf16<8><<<gGemm, TB, 0, stream>>>(hcur, gat_W + (size_t)r * WSZ,
                                                    nullptr, hn /*f*/, N, 0);
        gat_el_er<<<gN4, TB, 0, stream>>>(hn, gat_al + (size_t)r * DHID,
                                          gat_ar + (size_t)r * DHID, el, er, N);
        fill_f32<<<gN4, TB, 0, stream>>>(mx, -1e30f, (size_t)N * 4);
        fill_f32<<<gN4, TB, 0, stream>>>(ssum, 0.f, (size_t)N * 4);
        gat_edge_max<<<gEdge4, TB, 0, stream>>>(el, er, sr, dr, mx, E);
        gat_edge_sum<<<gEdge4, TB, 0, stream>>>(el, er, sr, dr, mx, ssum, E);
        gat_scatter<<<gEdgeW, TB, 0, stream>>>(hn, el, er, sr, dr, mx, ssum, acc, E);
    }
    gat_finalize<<<gElem, TB, 0, stream>>>(acc, gat_b, hcur, NF);

    // ---------------- Final linear: out = h @ lin_W + lin_b ----------------
    gemm_wmma_bf16<4><<<gGemm, TB, 0, stream>>>(hcur, lin_W, lin_b, out, N, 0);
}
